// Attention_66803921322221
// MI455X (gfx1250) — compile-verified, matrix operand order checked by reading, not run on hardware
//
#include <hip/hip_runtime.h>
#include <hip/hip_bf16.h>

typedef __attribute__((ext_vector_type(16))) _Float16 v16h;
typedef __attribute__((ext_vector_type(8)))  _Float16 v8h;
typedef __attribute__((ext_vector_type(8)))  float    v8f;

union V16 { v16h v; v8h h[2]; };

#define WMMA(a, b, c) __builtin_amdgcn_wmma_f32_16x16x32_f16( \
    false, (a), false, (b), (short)0, (c), false, false)

static constexpr int BATCH = 2;
static constexpr int SEQ   = 2048;
static constexpr int DIM   = 768;
static constexpr int HEADS = 12;
static constexpr int HD    = 64;
static constexpr int MROWS = BATCH * SEQ;   // 4096
static constexpr int QKVN  = 3 * DIM;       // 2304

// ---------------------------------------------------------------------------
// CDNA5 async global->LDS copy (ASYNCcnt-tracked, no VGPR data path).
// LDS byte address for the VGPR operand = low 32 bits of the generic pointer
// (flat LDS aperture).
// ---------------------------------------------------------------------------
__device__ __forceinline__ void async_g2l_b128(const _Float16* lds,
                                               const _Float16* g) {
  unsigned l = (unsigned)(unsigned long long)lds;
  unsigned long long ga = (unsigned long long)g;
  asm volatile("global_load_async_to_lds_b128 %0, %1, off"
               :: "v"(l), "v"(ga) : "memory");
}
__device__ __forceinline__ void wait_async0() {
#if __has_builtin(__builtin_amdgcn_s_wait_asynccnt)
  __builtin_amdgcn_s_wait_asynccnt(0);
#else
  asm volatile("s_wait_asynccnt 0" ::: "memory");
#endif
}

// ---------------------------------------------------------------------------
// f32 -> f16 conversion
// ---------------------------------------------------------------------------
__global__ void cvt_f32_to_f16(const float* __restrict__ in,
                               _Float16* __restrict__ out, int n) {
  int i = blockIdx.x * blockDim.x + threadIdx.x;
  int stride = gridDim.x * blockDim.x;
  for (; i < n; i += stride) out[i] = (_Float16)in[i];
}

// ---------------------------------------------------------------------------
// Shared 64x64 GEMM: C(64x64,f32) += A(64xK,f16) * B(Kx64,f16)
// 4 waves; wave w owns rows [w*16, w*16+16), 64 cols as 4 accumulators.
// Double-buffered LDS (indexed 2-D shared arrays -> stays in addrspace(3));
// A-tile staged by async copy, B-tile staged transposed.
// ---------------------------------------------------------------------------
typedef _Float16 Tile2[2][64 * 32];

__device__ __forceinline__ void gemm_stage(const _Float16* __restrict__ A, int lda,
                                           const _Float16* __restrict__ B, int ldb,
                                           int m0, int n0, int k0,
                                           Tile2& As, Tile2& Bt, int buf) {
  const int tid = threadIdx.x;
  // A tile 64x32 f16 row-major: 2 async b128 per thread
  {
    int row = tid >> 1, cb = (tid & 1) * 16;
    const _Float16* g = A + (size_t)(m0 + row) * lda + k0 + cb;
    async_g2l_b128(&As[buf][row * 32 + cb], g);
    async_g2l_b128(&As[buf][row * 32 + cb + 8], g + 8);
  }
  // B tile 32 k-rows x 64 cols: store TRANSPOSED -> Bt[buf][n][k]
  {
    int kr = tid >> 2, nb = (tid & 3) * 16;
    const _Float16* g = B + (size_t)(k0 + kr) * ldb + n0 + nb;
    v8h b0 = *(const v8h*)g, b1 = *(const v8h*)(g + 8);
#pragma unroll
    for (int i = 0; i < 8; i++) {
      Bt[buf][(nb + i) * 32 + kr]     = b0[i];
      Bt[buf][(nb + 8 + i) * 32 + kr] = b1[i];
    }
  }
}

__device__ __forceinline__ void gemm64x64(const _Float16* __restrict__ A, int lda,
                                          const _Float16* __restrict__ B, int ldb,
                                          int m0, int n0, int K,
                                          Tile2& As, Tile2& Bt, v8f acc[4]) {
  const int tid = threadIdx.x, wave = tid >> 5, lane = tid & 31;
  const int ln = lane & 15, hf = lane >> 4;
  gemm_stage(A, lda, B, ldb, m0, n0, 0, As, Bt, 0);
  wait_async0();
  __syncthreads();
  const int nsteps = K / 32;
  for (int s = 0; s < nsteps; ++s) {
    const int buf = s & 1;
    if (s + 1 < nsteps)  // stream next tile into the alternate buffer
      gemm_stage(A, lda, B, ldb, m0, n0, (s + 1) * 32, As, Bt, buf ^ 1);
    // batch all fragment loads, then burst the WMMAs
    V16 a, bb[4];
    const int abase = (wave * 16 + ln) * 32;
    a.h[0] = *(const v8h*)&As[buf][abase + hf * 8];
    a.h[1] = *(const v8h*)&As[buf][abase + 16 + hf * 8];
#pragma unroll
    for (int c = 0; c < 4; c++) {
      const int bbase = (c * 16 + ln) * 32;
      bb[c].h[0] = *(const v8h*)&Bt[buf][bbase + hf * 8];
      bb[c].h[1] = *(const v8h*)&Bt[buf][bbase + 16 + hf * 8];
    }
#pragma unroll
    for (int c = 0; c < 4; c++) acc[c] = WMMA(a.v, bb[c].v, acc[c]);
    wait_async0();
    __syncthreads();
  }
}

// ---------------------------------------------------------------------------
// Kernel 1: qkv = x @ w_qkv ; scatter to q/k/v [B,H,SEQ,HD] f16, q pre-scaled
// ---------------------------------------------------------------------------
__global__ __launch_bounds__(128) void qkv_gemm(const _Float16* __restrict__ Xh,
                                                const _Float16* __restrict__ Wh,
                                                _Float16* __restrict__ Qd,
                                                _Float16* __restrict__ Kd,
                                                _Float16* __restrict__ Vd) {
  __shared__ __align__(16) _Float16 As[2][64 * 32];
  __shared__ __align__(16) _Float16 Bt[2][64 * 32];
  const int NB = QKVN / 64;  // 36
  const int m0 = (blockIdx.x / NB) * 64, n0 = (blockIdx.x % NB) * 64;
  const int tid = threadIdx.x, wave = tid >> 5, lane = tid & 31;
  const int ln = lane & 15, hf = lane >> 4;
  v8f acc[4] = {};
  gemm64x64(Xh, DIM, Wh, QKVN, m0, n0, DIM, As, Bt, acc);
#pragma unroll
  for (int c = 0; c < 4; c++) {
    int cg = n0 + c * 16 + ln;           // output column in [0, 2304)
    int s = cg / DIM;                    // 0=q 1=k 2=v
    int rem = cg - s * DIM;
    int h = rem >> 6, d = rem & 63;
    _Float16* dst = (s == 0) ? Qd : ((s == 1) ? Kd : Vd);
    float mul = (s == 0) ? 0.125f : 1.0f;  // fold 1/sqrt(64) into q
#pragma unroll
    for (int r = 0; r < 8; r++) {
      int m = m0 + wave * 16 + r + hf * 8;  // C-layout rows r / r+8
      int b = m >> 11, n = m & (SEQ - 1);
      dst[((size_t)(b * HEADS + h) * SEQ + n) * HD + d] = (_Float16)(acc[c][r] * mul);
    }
  }
}

// ---------------------------------------------------------------------------
// Kernel 2: fused masked-softmax attention, one (b,h) x 64 query rows / block
// K tile async-staged row-major; V tile transposed; both double-buffered.
// ---------------------------------------------------------------------------
typedef _Float16 KTile2[2][32 * 64];
typedef _Float16 VTile2[2][64 * 32];

__device__ __forceinline__ void attn_stage(const _Float16* __restrict__ kp,
                                           const _Float16* __restrict__ vp,
                                           int kt, KTile2& Ks, VTile2& Vt, int buf) {
  const int tid = threadIdx.x;
  const int row = tid >> 2, cb = (tid & 3) * 16;
  const _Float16* gk = kp + (size_t)(kt + row) * HD + cb;
  async_g2l_b128(&Ks[buf][row * 64 + cb], gk);
  async_g2l_b128(&Ks[buf][row * 64 + cb + 8], gk + 8);
  const _Float16* gv = vp + (size_t)(kt + row) * HD + cb;
  v8h v0 = *(const v8h*)gv, v1 = *(const v8h*)(gv + 8);
#pragma unroll
  for (int i = 0; i < 8; i++) {
    Vt[buf][(cb + i) * 32 + row]     = v0[i];
    Vt[buf][(cb + 8 + i) * 32 + row] = v1[i];
  }
}

__global__ __launch_bounds__(128) void attn_kernel(const _Float16* __restrict__ Qd,
                                                   const _Float16* __restrict__ Kd,
                                                   const _Float16* __restrict__ Vd,
                                                   const int* __restrict__ mask,
                                                   _Float16* __restrict__ Ah) {
  __shared__ __align__(16) _Float16 Ks[2][32 * 64];   // K tile, row-major
  __shared__ __align__(16) _Float16 Vt[2][64 * 32];   // V tile, transposed
  __shared__ __align__(16) _Float16 Ps[4][16 * 32];   // per-wave P staging
  const int QB = SEQ / 64;
  const int bh = blockIdx.x / QB, qb = blockIdx.x % QB;
  const int b = bh / HEADS, h = bh % HEADS;
  const int tid = threadIdx.x, wave = tid >> 5, lane = tid & 31;
  const int ln = lane & 15, hf = lane >> 4;

  const _Float16* q  = Qd + (size_t)bh * SEQ * HD;
  const _Float16* kp = Kd + (size_t)bh * SEQ * HD;
  const _Float16* vp = Vd + (size_t)bh * SEQ * HD;

  // Q fragments (16x64) in registers, A-layout
  const int qrow = qb * 64 + wave * 16 + ln;
  V16 qa0, qa1;
  qa0.h[0] = *(const v8h*)&q[qrow * HD + hf * 8];
  qa0.h[1] = *(const v8h*)&q[qrow * HD + 16 + hf * 8];
  qa1.h[0] = *(const v8h*)&q[qrow * HD + 32 + hf * 8];
  qa1.h[1] = *(const v8h*)&q[qrow * HD + 48 + hf * 8];

  v8f o[4] = {};
  float den = 0.0f;

  attn_stage(kp, vp, 0, Ks, Vt, 0);
  wait_async0();
  __syncthreads();

  const int nsteps = SEQ / 32;
  for (int s = 0; s < nsteps; ++s) {
    const int buf = s & 1;
    const int kt = s * 32;
    if (s + 1 < nsteps)  // stream next K/V tile while computing this one
      attn_stage(kp, vp, kt + 32, Ks, Vt, buf ^ 1);

    const float m0v = (float)mask[b * SEQ + kt + ln];
    const float m1v = (float)mask[b * SEQ + kt + 16 + ln];

    // --- S = Q K^T (pre-scaled); two 16-key subtiles, K-dim 64 = 2 WMMAs ---
    V16 kb[2][2];
#pragma unroll
    for (int s2 = 0; s2 < 2; s2++) {
      const int krb = (s2 * 16 + ln) * 64;
      kb[s2][0].h[0] = *(const v8h*)&Ks[buf][krb + hf * 8];
      kb[s2][0].h[1] = *(const v8h*)&Ks[buf][krb + 16 + hf * 8];
      kb[s2][1].h[0] = *(const v8h*)&Ks[buf][krb + 32 + hf * 8];
      kb[s2][1].h[1] = *(const v8h*)&Ks[buf][krb + 48 + hf * 8];
    }
    v8f sacc[2];
#pragma unroll
    for (int s2 = 0; s2 < 2; s2++) {
      v8f t = {};
      t = WMMA(qa0.v, kb[s2][0].v, t);
      t = WMMA(qa1.v, kb[s2][1].v, t);
      sacc[s2] = t;
    }

    // --- P = exp(S)*mask (faithful unstable softmax), C-layout -> LDS ---
#pragma unroll
    for (int s2 = 0; s2 < 2; s2++) {
      float mm = s2 ? m1v : m0v;
#pragma unroll
      for (int r = 0; r < 8; r++) {
        float p = __expf(sacc[s2][r]) * mm;
        Ps[wave][(r + hf * 8) * 32 + s2 * 16 + ln] = (_Float16)p;
      }
    }
    asm volatile("s_wait_dscnt 0" ::: "memory");  // P staging visible in-wave

    // --- reload P in A-layout for the PV matmul ---
    V16 pa;
    pa.h[0] = *(const v8h*)&Ps[wave][ln * 32 + hf * 8];
    pa.h[1] = *(const v8h*)&Ps[wave][ln * 32 + 16 + hf * 8];

    // row denominators: per-lane partial over 16 keys + xor-16 completes 32
    float dsum = 0.0f;
#pragma unroll
    for (int i = 0; i < 16; i++) dsum += (float)pa.v[i];
    dsum += __shfl_xor(dsum, 16, 32);
    den += dsum;

    // --- O += P V : 4 d-chunks, K-dim 32 = 1 WMMA each ---
    V16 vb[4];
#pragma unroll
    for (int c = 0; c < 4; c++) {
      const int vcb = (c * 16 + ln) * 32;
      vb[c].h[0] = *(const v8h*)&Vt[buf][vcb + hf * 8];
      vb[c].h[1] = *(const v8h*)&Vt[buf][vcb + 16 + hf * 8];
    }
#pragma unroll
    for (int c = 0; c < 4; c++) o[c] = WMMA(pa.v, vb[c].v, o[c]);

    wait_async0();
    __syncthreads();
  }

  // --- normalize and store f16 attention output [B*SEQ, DIM] ---
  const int mbase = qb * 64 + wave * 16;
#pragma unroll
  for (int c = 0; c < 4; c++) {
    int d = c * 16 + ln;
#pragma unroll
    for (int r = 0; r < 8; r++) {
      int M = r + hf * 8;
      float dr = __shfl(den, M, 32);   // den lives in lane M (row M)
      float val = o[c][r] / dr;
      Ah[((size_t)(b * SEQ + mbase + M)) * DIM + h * HD + d] = (_Float16)val;
    }
  }
}

// ---------------------------------------------------------------------------
// Kernel 3: out = attn @ w_proj + b_proj  (fp32 output)
// ---------------------------------------------------------------------------
__global__ __launch_bounds__(128) void proj_gemm(const _Float16* __restrict__ Ah,
                                                 const _Float16* __restrict__ Wp,
                                                 const float* __restrict__ bias,
                                                 float* __restrict__ out) {
  __shared__ __align__(16) _Float16 As[2][64 * 32];
  __shared__ __align__(16) _Float16 Bt[2][64 * 32];
  const int NB = DIM / 64;
  const int m0 = (blockIdx.x / NB) * 64, n0 = (blockIdx.x % NB) * 64;
  const int tid = threadIdx.x, wave = tid >> 5, lane = tid & 31;
  const int ln = lane & 15, hf = lane >> 4;
  v8f acc[4] = {};
  gemm64x64(Ah, DIM, Wp, DIM, m0, n0, DIM, As, Bt, acc);
#pragma unroll
  for (int c = 0; c < 4; c++) {
    int cg = n0 + c * 16 + ln;
    float bv = bias[cg];
#pragma unroll
    for (int r = 0; r < 8; r++) {
      int m = m0 + wave * 16 + r + hf * 8;
      out[(size_t)m * DIM + cg] = acc[c][r] + bv;
    }
  }
}

// ---------------------------------------------------------------------------
// Launch
// ---------------------------------------------------------------------------
extern "C" void kernel_launch(void* const* d_in, const int* in_sizes, int n_in,
                              void* d_out, int out_size, void* d_ws, size_t ws_size,
                              hipStream_t stream) {
  const float* x      = (const float*)d_in[0];
  const int*   mask   = (const int*)d_in[1];
  const float* w_qkv  = (const float*)d_in[2];
  const float* w_proj = (const float*)d_in[3];
  const float* b_proj = (const float*)d_in[4];
  float* out = (float*)d_out;

  // workspace layout (f16 elems); total ~36.2 MB
  _Float16* ws = (_Float16*)d_ws;
  size_t off = 0;
  _Float16* xh  = ws + off; off += (size_t)MROWS * DIM;
  _Float16* wqh = ws + off; off += (size_t)DIM * QKVN;
  _Float16* wph = ws + off; off += (size_t)DIM * DIM;
  _Float16* qh  = ws + off; off += (size_t)BATCH * HEADS * SEQ * HD;
  _Float16* kh  = ws + off; off += (size_t)BATCH * HEADS * SEQ * HD;
  _Float16* vh  = ws + off; off += (size_t)BATCH * HEADS * SEQ * HD;
  _Float16* ah  = ws + off; off += (size_t)MROWS * DIM;

  cvt_f32_to_f16<<<2048, 256, 0, stream>>>(x, xh, MROWS * DIM);
  cvt_f32_to_f16<<<2048, 256, 0, stream>>>(w_qkv, wqh, DIM * QKVN);
  cvt_f32_to_f16<<<1024, 256, 0, stream>>>(w_proj, wph, DIM * DIM);

  qkv_gemm<<<(MROWS / 64) * (QKVN / 64), 128, 0, stream>>>(xh, wqh, qh, kh, vh);
  attn_kernel<<<BATCH * HEADS * (SEQ / 64), 128, 0, stream>>>(qh, kh, vh, mask, ah);
  proj_gemm<<<(MROWS / 64) * (DIM / 64), 128, 0, stream>>>(ah, wph, b_proj, out);
}
